// SpringEquationNN_70102456205450
// MI455X (gfx1250) — matrix operand order repaired and checked by reading, not compiled, not run on hardware
//
#include <hip/hip_runtime.h>

typedef __attribute__((ext_vector_type(16))) _Float16 v16h;
typedef __attribute__((ext_vector_type(2)))  _Float16 h2;
typedef __attribute__((ext_vector_type(8)))  float    v8f;
typedef __attribute__((ext_vector_type(8)))  int      v8i;

#define HID 20
#define SWAPX16 0x401f   // ds_swizzle group-of-32: xor=0x10, or=0, and=0x1f

struct MLPParams {
  const float* t;
  const float* W0; const float* b0;
  const float* Wh[6]; const float* bh[6];
  const float* W7; const float* b7;
  float* out;
  int n;
};

// silu(x) = x*sigmoid(x) = h + h*tanh(h), h = 0.5x  (1 trans + 2 VALU, packs well)
__device__ __forceinline__ float silu_f(float x) {
  const float h = 0.5f * x;
  float th;
  asm("v_tanh_f32 %0, %1" : "=v"(th) : "v"(h));
  return fmaf(h, th, h);
}

// Feature-major formulation: activations live as Ht[k][point] in the WMMA
// B-operand layout (lane = point, K = features, padded 20->32 with row 20
// pinned to 1.0 so biases ride in weight column k=20).  Every layer 0..6 is
//   D(t) = A_w[l][t] x B,  t in {0,1} covering out-features 16t..16t+15.
// D -> next B needs only a half-wave swap (4x ds_swizzle SWAPX16) + selects.
// Padding rows (features 21..31) are exact zeros end-to-end, so only the
// 4 real rows of tile 1 (features 16..19) are activated/packed.
__global__ __launch_bounds__(256) void pinn_mlp_wmma(MLPParams p) {
  const int tid  = threadIdx.x;
  const int lane = tid & 31;
  const int half = lane >> 4;   // 0 or 1
  const int l15  = lane & 15;   // point / row-in-tile index

  // ---- Weight-stationary A fragments for layers 0..6 (112 VGPRs) ----
  // A f16 fragment, k order used here for A and B: lane L row M = L%16, element j -> K = j + 16*(L>=16).
  // Value = W[l][n][k] for k<20, bias b[l][n] at k==20, else 0.
  v16h af[7][2];
  #pragma unroll
  for (int l = 0; l < 7; ++l) {
    const float* W = (l == 0) ? p.W0 : p.Wh[l - 1];
    const float* b = (l == 0) ? p.b0 : p.bh[l - 1];
    #pragma unroll
    for (int t = 0; t < 2; ++t) {
      const int n = 16 * t + l15;
      #pragma unroll
      for (int j = 0; j < 16; ++j) {
        const int k = j + 16 * half;
        float v = 0.0f;
        if (n < HID) {
          if (l == 0)       v = (k == 0) ? W[n] : ((k == 20) ? b[n] : 0.0f);
          else if (k < HID) v = W[n * HID + k];
          else if (k == 20) v = b[n];
        }
        af[l][t][j] = (_Float16)v;
      }
    }
  }

  // ---- Output-layer weights per lane: s0 covers feats r+8*half, s1 feats 16..19 ----
  float w7a[8], w7b[4];
  #pragma unroll
  for (int r = 0; r < 8; ++r) {
    const int ka = r + 8 * half;
    w7a[r] = (ka < HID) ? p.W7[ka] : 0.0f;
  }
  #pragma unroll
  for (int r = 0; r < 4; ++r) {
    const int kb = 16 + r + 8 * half;
    w7b[r] = (kb < HID) ? p.W7[kb] : 0.0f;
  }
  const float bias7 = p.b7[0];

  const v8f czero = {};
  const int tiles  = (p.n + 15) >> 4;
  const int gwave  = blockIdx.x * 8 + (tid >> 5);
  const int nwaves = gridDim.x * 8;

  for (int tile = gwave; tile < tiles; tile += nwaves) {
    const int base = tile << 4;
    int tidx = base + l15;
    if (tidx >= p.n) tidx = p.n - 1;
    const float tval = p.t[tidx];

    // ---- Initial Ht = [t ; 1] in B layout: lane holds its point's features ----
    // half0 dword0 = (t, 0) -> feats 0,1 ; half1 dword2 = (1.0h, 0) -> feats 20,21.
    v8i bv = {};
    h2 t0; t0.x = (_Float16)tval; t0.y = (_Float16)0.0f;
    bv[0] = half ? 0 : __builtin_bit_cast(int, t0);
    bv[2] = half ? 0x3C00 : 0;

    float res = 0.0f;

    #pragma unroll
    for (int l = 0; l < 7; ++l) {
      const v16h bmat = __builtin_bit_cast(v16h, bv);
      // D element (M = r + 8*half [out-feature], N = L%16 [point]) in VGPR r.
      v8f d0 = __builtin_amdgcn_wmma_f32_16x16x32_f16(
          false, af[l][0], false, bmat, (short)0, czero, false, false);
      v8f d1 = __builtin_amdgcn_wmma_f32_16x16x32_f16(
          false, af[l][1], false, bmat, (short)0, czero, false, false);

      float s0[8], s1[4];
      #pragma unroll
      for (int r = 0; r < 8; ++r) s0[r] = silu_f(d0[r]);
      #pragma unroll
      for (int r = 0; r < 4; ++r) s1[r] = silu_f(d1[r]);  // feats 16..19 (half1: zeros)

      if (l < 6) {
        // Pack to f16 pairs: p0[q] = feats (2q+8*half, +1), p1[q] = 16 + same.
        int p0[4], p1[4];
        #pragma unroll
        for (int q = 0; q < 4; ++q) {
          h2 a; a.x = (_Float16)s0[2 * q]; a.y = (_Float16)s0[2 * q + 1];
          p0[q] = __builtin_bit_cast(int, a);
        }
        {
          h2 b0h; b0h.x = (_Float16)s1[0]; b0h.y = (_Float16)s1[1];
          h2 b1h; b1h.x = (_Float16)s1[2]; b1h.y = (_Float16)s1[3];
          p1[0] = __builtin_bit_cast(int, b0h);
          p1[1] = __builtin_bit_cast(int, b1h);
          p1[2] = 0;   // feats 20,21: receiver re-pins 20 to 1.0, 21 is zero
          p1[3] = 0;   // feats 22,23: exact zeros
        }
        // Half-wave exchange: half0 wants partner's p0 (feats 8-15),
        // half1 wants partner's p1 (feats 16-23). Send what partner needs.
        #pragma unroll
        for (int q = 0; q < 4; ++q) {
          const int send = half ? p0[q] : p1[q];
          const int recv = __builtin_amdgcn_ds_swizzle(send, SWAPX16);
          bv[q]     = half ? recv  : p0[q];   // feats 16*half + 2q, +1
          bv[4 + q] = half ? p1[q] : recv;    // feats 16*half + 8 + 2q, +1
        }
        // Re-pin constant-one row: feature 20 = half1, dword 2, low half.
        bv[2] = half ? 0x3C00 : bv[2];
      } else {
        // ---- Layer 7 (20->1): per-point dot from registers ----
        float part = 0.0f;
        #pragma unroll
        for (int r = 0; r < 8; ++r) part = fmaf(s0[r], w7a[r], part);
        #pragma unroll
        for (int r = 0; r < 4; ++r) part = fmaf(s1[r], w7b[r], part);
        const int pr = __builtin_amdgcn_ds_swizzle(__builtin_bit_cast(int, part), SWAPX16);
        res = part + __builtin_bit_cast(float, pr) + bias7;
      }
    }

    const int oidx = base + l15;
    if (lane < 16 && oidx < p.n) p.out[oidx] = res;
  }
}

extern "C" void kernel_launch(void* const* d_in, const int* in_sizes, int n_in,
                              void* d_out, int out_size, void* d_ws, size_t ws_size,
                              hipStream_t stream) {
  (void)n_in; (void)d_ws; (void)ws_size; (void)out_size;
  MLPParams prm;
  prm.t  = (const float*)d_in[0];
  prm.W0 = (const float*)d_in[1];
  prm.b0 = (const float*)d_in[2];
  for (int i = 0; i < 6; ++i) {
    prm.Wh[i] = (const float*)d_in[3 + 2 * i];
    prm.bh[i] = (const float*)d_in[4 + 2 * i];
  }
  prm.W7  = (const float*)d_in[15];
  prm.b7  = (const float*)d_in[16];
  prm.out = (float*)d_out;
  prm.n   = in_sizes[0];   // N (t is [N,1])

  // 1024 blocks x 8 waves = 8192 waves -> 8 tiles of 16 points per wave.
  pinn_mlp_wmma<<<1024, 256, 0, stream>>>(prm);
}
